// IDSSM_58351425683908
// MI455X (gfx1250) — matrix operand order repaired and checked by reading, not compiled, not run on hardware
//
#include <hip/hip_runtime.h>
#include <hip/hip_bf16.h>

#define B_TOTAL 65536
#define NSENS   32
#define LATENT  32
#define HIDDEN  64

typedef __attribute__((ext_vector_type(16))) _Float16 v16h;
typedef __attribute__((ext_vector_type(8)))  float    v8f;

__device__ __forceinline__ float fast_rcp(float x) {
    return __builtin_amdgcn_rcpf(x);            // raw v_rcp_f32, ~1 ulp
}

__device__ __forceinline__ float fast_tanh(float x) {
    // tanh(x) = 1 - 2/(e^{2x}+1); exp + add + rcp + fma, saturates at +-1
    float e = __expf(2.0f * x);
    return fmaf(-2.0f, fast_rcp(e + 1.0f), 1.0f);
}

// ---------------------------------------------------------------------------
// Kernel 0: derived constants + WMMA-layout weight packing
// ws float region: [0]=c1 [1]=c2 [2..17]=q[16] [32..95]=exp(m1_w)[64] (16B-aligned)
// b2: exp(m2_w) as B-operand frags: (tile t=0..3)(chunk c=0..1)(lane 0..31)(16 halves)
// b3: exp(m3_w) as B-operand frags: (tile t=0..1)(chunk c=0..1)(lane 0..31)(16 halves)
// B[k][n] per lane: lane<16 -> col n=t*16+lane,    K = c*32 + {0..7, 16..23}
//                   lane>=16 -> col n=t*16+lane-16, K = c*32 + {8..15, 24..31}
// ---------------------------------------------------------------------------
__global__ void __launch_bounds__(256)
prep_kernel(const float* __restrict__ W_gat, const float* __restrict__ a_gat,
            const float* __restrict__ p1_w,  const float* __restrict__ m1_w,
            const float* __restrict__ m2_w,  const float* __restrict__ m3_w,
            float* __restrict__ consts, _Float16* __restrict__ b2,
            _Float16* __restrict__ b3) {
    int tid = threadIdx.x;
    if (tid == 0) {
        float c1 = 0.f, c2 = 0.f;
        for (int d = 0; d < LATENT; ++d) {
            c1 += W_gat[d] * a_gat[d];
            c2 += W_gat[d] * a_gat[LATENT + d];
        }
        consts[0] = c1;
        consts[1] = c2;
    }
    if (tid < 16) {
        float q = 0.f;
        for (int d = 0; d < LATENT; ++d) q += p1_w[tid * LATENT + d] * W_gat[d];
        consts[2 + tid] = q;
    }
    if (tid < HIDDEN) consts[32 + tid] = __expf(m1_w[tid]);

    // pack exp(m2_w): B[k][n] = exp(m2_w[n*64+k]), 4 N-tiles x 2 K-chunks
    for (int e = tid; e < 4096; e += 256) {
        int t = e >> 10, rem = e & 1023;
        int c = rem >> 9; int r2 = rem & 511;
        int L = r2 >> 4, v = r2 & 15;
        int n = t * 16 + (L & 15);
        int k = c * 32 + ((v >> 3) << 4) + ((L >> 4) << 3) + (v & 7);
        b2[e] = (_Float16)__expf(m2_w[n * HIDDEN + k]);
    }
    // pack exp(m3_w): B[k][n] = exp(m3_w[n*64+k]), 2 N-tiles x 2 K-chunks
    for (int e = tid; e < 2048; e += 256) {
        int t = e >> 10, rem = e & 1023;
        int c = rem >> 9; int r2 = rem & 511;
        int L = r2 >> 4, v = r2 & 15;
        int n = t * 16 + (L & 15);
        int k = c * 32 + ((v >> 3) << 4) + ((L >> 4) << 3) + (v & 7);
        b3[e] = (_Float16)__expf(m3_w[n * HIDDEN + k]);
    }
}

// ---------------------------------------------------------------------------
// Kernel 1: encoder (GAT + attention pooling), one wave per batch, lane = sensor
// ---------------------------------------------------------------------------
__global__ void __launch_bounds__(256)
enc_kernel(const float* __restrict__ x_sensors, const float* __restrict__ W_gat,
           const float* __restrict__ p1_b, const float* __restrict__ p2_w,
           const float* __restrict__ p2_b, const float* __restrict__ consts,
           float* __restrict__ z_enc) {
    int lane = threadIdx.x & 31;
    int b    = (blockIdx.x * blockDim.x + threadIdx.x) >> 5;

    float c1 = consts[0], c2 = consts[1];
    float x  = x_sensors[b * NSENS + lane];
    float s1 = c1 * x;          // s_i contribution (this lane's row)
    float t2 = c2 * x;          // s_j contribution (this lane's column)

    // row max: lrelu monotone -> max_j lrelu(s1_i + t2_j) = lrelu(s1_i + max_j t2_j)
    float mx = t2;
    for (int off = 16; off; off >>= 1) mx = fmaxf(mx, __shfl_xor(mx, off));
    float vv   = s1 + mx;
    float mrow = fmaxf(vv, 0.2f * vv);

    // softmax over j + alpha @ x, streamed; t2_j = c2*x_j so one shuffle per j
    float sum = 0.f, g = 0.f;
    for (int j = 0; j < NSENS; ++j) {
        float xj = __shfl(x, j);
        float v  = fmaf(c2, xj, s1);
        v = fmaxf(v, 0.2f * v);              // leaky_relu(v, 0.2)
        float p = __expf(v - mrow);
        sum += p;
        g = fmaf(p, xj, g);
    }
    g *= fast_rcp(sum);                      // g_i = sum_j alpha_ij x_j

    // attention pooling score u_i = p2_w . tanh(g_i * q + p1_b) + p2_b
    float u = p2_b[0];
    for (int k = 0; k < 16; ++k) {
        float tk = fast_tanh(fmaf(g, consts[2 + k], p1_b[k]));
        u = fmaf(tk, p2_w[k], u);
    }
    // beta = softmax_i(u); G = sum_i beta_i g_i
    float mu = u;
    for (int off = 16; off; off >>= 1) mu = fmaxf(mu, __shfl_xor(mu, off));
    float pe = __expf(u - mu);
    float se = pe, gw = pe * g;
    for (int off = 16; off; off >>= 1) {
        se += __shfl_xor(se, off);
        gw += __shfl_xor(gw, off);
    }
    float G = gw * fast_rcp(se);
    z_enc[b * LATENT + lane] = G * W_gat[lane];   // z_enc rank-1: G_b * w_d
}

// ---------------------------------------------------------------------------
// Kernel 2: decoder MLP via WMMA. One wave handles 16 batch rows.
// h1 computed straight into A-fragments (each lane owns one M-row, 4x8 K-runs);
// h2 relayout (D: N-per-lane -> A: M-per-lane) goes through LDS.
// ---------------------------------------------------------------------------
union AF { v16h v; _Float16 h[16]; uint4 q[2]; };

__device__ __forceinline__ void load_afrag_lds(const _Float16* base, AF* a,
                                               int mrow, int hi) {
    // A 16x32 f16 layout: lane<16 holds K {c*32+0..7, c*32+16..23}; lane>=16: +8
    for (int c = 0; c < 2; ++c) {
        const _Float16* p = base + mrow * HIDDEN + c * 32 + hi * 8;
        a[c].q[0] = *(const uint4*)(p);
        a[c].q[1] = *(const uint4*)(p + 16);
    }
}

__global__ void __launch_bounds__(256)
dec_kernel(const float* __restrict__ state_x, const float* __restrict__ m1_b,
           const float* __restrict__ m2_b,   const float* __restrict__ m3_b,
           const float* __restrict__ consts, const _Float16* __restrict__ b2,
           const _Float16* __restrict__ b3,  float* __restrict__ z_dec) {
    __shared__ __align__(16) _Float16 s_h2[8][16 * HIDDEN];

    int lane = threadIdx.x & 31;
    int w    = threadIdx.x >> 5;
    int b0   = (blockIdx.x * 8 + w) * 16;     // 16 batches per wave
    int mrow = lane & 15;
    int hi   = lane >> 4;

    // h1 A-fragments in-register: lane covers row m=mrow,
    // K = c*32 + gsel*16 + hi*8 + 0..7  for c,gsel in {0,1}
    AF a1[2];
    {
        float sm = state_x[b0 + mrow];
        const float* E1 = consts + 32;
        for (int c = 0; c < 2; ++c) {
            for (int gsel = 0; gsel < 2; ++gsel) {
                int k0 = c * 32 + gsel * 16 + hi * 8;
                float ev[8], bv[8];
                *(float4*)&ev[0] = *(const float4*)(E1 + k0);
                *(float4*)&ev[4] = *(const float4*)(E1 + k0 + 4);
                *(float4*)&bv[0] = *(const float4*)(m1_b + k0);
                *(float4*)&bv[4] = *(const float4*)(m1_b + k0 + 4);
                for (int e = 0; e < 8; ++e) {
                    a1[c].h[gsel * 8 + e] =
                        (_Float16)fast_tanh(fmaf(sm, ev[e], bv[e]));
                }
            }
        }
    }

    // h2 = tanh(h1 @ exp(m2_w)^T + m2_b): 4 N-tiles x 2 K-chunk WMMA chain
    for (int t = 0; t < 4; ++t) {
        v8f c = {};
        for (int ch = 0; ch < 2; ++ch) {
            AF bf;
            const _Float16* bb = b2 + ((t * 2 + ch) * 32 + lane) * 16;
            bf.q[0] = *(const uint4*)(bb);
            bf.q[1] = *(const uint4*)(bb + 8);
            c = __builtin_amdgcn_wmma_f32_16x16x32_f16(
                    false, a1[ch].v, false, bf.v, (short)0, c, false, false);
        }
        int j  = t * 16 + mrow;               // D layout: N fixed per lane
        float bj = m2_b[j];
        for (int r = 0; r < 8; ++r) {
            int m = r + hi * 8;
            s_h2[w][m * HIDDEN + j] = (_Float16)fast_tanh(c[r] + bj);
        }
    }
    __syncthreads();

    AF a2[2];
    load_afrag_lds(&s_h2[w][0], a2, mrow, hi);

    // z_dec = h2 @ exp(m3_w)^T + m3_b: 2 N-tiles x 2 K-chunk WMMA chain
    for (int t = 0; t < 2; ++t) {
        v8f c = {};
        for (int ch = 0; ch < 2; ++ch) {
            AF bf;
            const _Float16* bb = b3 + ((t * 2 + ch) * 32 + lane) * 16;
            bf.q[0] = *(const uint4*)(bb);
            bf.q[1] = *(const uint4*)(bb + 8);
            c = __builtin_amdgcn_wmma_f32_16x16x32_f16(
                    false, a2[ch].v, false, bf.v, (short)0, c, false, false);
        }
        int n  = t * 16 + mrow;
        float bd = m3_b[n];
        for (int r = 0; r < 8; ++r) {
            int m = r + hi * 8;
            z_dec[(size_t)(b0 + m) * LATENT + n] = c[r] + bd;
        }
    }
}

extern "C" void kernel_launch(void* const* d_in, const int* in_sizes, int n_in,
                              void* d_out, int out_size, void* d_ws, size_t ws_size,
                              hipStream_t stream) {
    (void)in_sizes; (void)n_in; (void)out_size; (void)ws_size;
    const float* x_sensors = (const float*)d_in[0];
    const float* state_x   = (const float*)d_in[1];
    const float* W_gat     = (const float*)d_in[2];
    const float* a_gat     = (const float*)d_in[3];
    const float* p1_w      = (const float*)d_in[4];
    const float* p1_b      = (const float*)d_in[5];
    const float* p2_w      = (const float*)d_in[6];
    const float* p2_b      = (const float*)d_in[7];
    const float* m1_w      = (const float*)d_in[8];
    const float* m1_b      = (const float*)d_in[9];
    const float* m2_w      = (const float*)d_in[10];
    const float* m2_b      = (const float*)d_in[11];
    const float* m3_w      = (const float*)d_in[12];
    const float* m3_b      = (const float*)d_in[13];

    float* z_enc = (float*)d_out;
    float* z_dec = z_enc + (size_t)B_TOTAL * LATENT;

    float*    consts = (float*)d_ws;
    _Float16* b2     = (_Float16*)((char*)d_ws + 512);
    _Float16* b3     = (_Float16*)((char*)d_ws + 512 + 8192);

    prep_kernel<<<1, 256, 0, stream>>>(W_gat, a_gat, p1_w, m1_w, m2_w, m3_w,
                                       consts, b2, b3);
    enc_kernel<<<B_TOTAL / 8, 256, 0, stream>>>(x_sensors, W_gat, p1_b, p2_w,
                                                p2_b, consts, z_enc);
    dec_kernel<<<B_TOTAL / (16 * 8), 256, 0, stream>>>(state_x, m1_b, m2_b, m3_b,
                                                       consts, b2, b3, z_dec);
}
